// CausalSelfAttention_48369921688055
// MI455X (gfx1250) — matrix operand order repaired and checked by reading, not compiled, not run on hardware
//
#include <hip/hip_runtime.h>
#include <hip/hip_bf16.h>

#define Bc 4
#define Tc 2048
#define Cc 1024
#define NHc 16
#define Dc 64

typedef __bf16 bf16_t;
typedef __attribute__((ext_vector_type(16))) __bf16 v16bf;
typedef __attribute__((ext_vector_type(8)))  __bf16 v8bf;
typedef __attribute__((ext_vector_type(8)))  float  v8f;

// ---------------- fragment loaders (bf16, global memory) ----------------
// A-matrix 16x32 (MxK), row-major source with leading dim ld.
// lane<16: M=lane, K halves {0..7, 16..23}; lane>=16: M=lane-16, K {8..15, 24..31}
__device__ __forceinline__ v16bf load_a_g(const bf16_t* __restrict__ base, int ld, int lane) {
    const int koff = (lane & 16) ? 8 : 0;
    const bf16_t* p = base + (size_t)(lane & 15) * ld + koff;
    v8bf lo = *(const v8bf*)(p);
    v8bf hi = *(const v8bf*)(p + 16);
    v16bf r;
#pragma unroll
    for (int i = 0; i < 8; ++i) { r[i] = lo[i]; r[i + 8] = hi[i]; }
    return r;
}

// B-matrix 32x16 (KxN). Source rows are the *columns* of B (e.g. a row of W for W^T GEMMs),
// contiguous along K with leading dim ld. lane<16: N=lane, K {0..7, 16..23}; lane>=16: N=lane-16, K {8..15, 24..31}
__device__ __forceinline__ v16bf load_b_g(const bf16_t* __restrict__ base, int ld, int lane) {
    const int koff = (lane & 16) ? 8 : 0;
    const bf16_t* p = base + (size_t)(lane & 15) * ld + koff;
    v8bf lo = *(const v8bf*)(p);
    v8bf hi = *(const v8bf*)(p + 16);
    v16bf r;
#pragma unroll
    for (int i = 0; i < 8; ++i) { r[i] = lo[i]; r[i + 8] = hi[i]; }
    return r;
}

__device__ __forceinline__ v8f wmma_bf16(v16bf a, v16bf b, v8f c) {
    return __builtin_amdgcn_wmma_f32_16x16x32_bf16(false, a, false, b, (short)0, c, false, false);
}

// ---------------- fp32 -> bf16 conversion ----------------
__global__ void cvt_bf16_kernel(const float* __restrict__ s, bf16_t* __restrict__ d, int n) {
    int i = blockIdx.x * blockDim.x + threadIdx.x;
    if (i < n) d[i] = (bf16_t)s[i];
}

// ---------------- QKV GEMM (+bias) with fused RoPE epilogue ----------------
// qkv = x @ w_attn^T + b_attn ; q,k get RoPE; q,k -> [B,NH,T,D]; v -> [B,NH,D,T] (transposed)
__global__ __launch_bounds__(128) void qkv_rope_kernel(
    const bf16_t* __restrict__ xb, const bf16_t* __restrict__ wa,
    const float* __restrict__ b_attn,
    const float* __restrict__ cosT, const float* __restrict__ sinT,
    bf16_t* __restrict__ qb, bf16_t* __restrict__ kbuf, bf16_t* __restrict__ vtb)
{
    const int lane = threadIdx.x & 31;
    const int wave = threadIdx.x >> 5;
    const int m0 = blockIdx.y * 64 + wave * 16;   // rows of [B*T]
    const int n0 = blockIdx.x * 64;               // cols of [3C], head-aligned
    const int K = Cc;

    v8f acc[4];
#pragma unroll
    for (int c = 0; c < 4; ++c)
#pragma unroll
        for (int i = 0; i < 8; ++i) acc[c][i] = 0.0f;

    const bf16_t* arow = xb + (size_t)m0 * K;
    const bf16_t* brow = wa + (size_t)n0 * K;
    for (int k0 = 0; k0 < K; k0 += 32) {
        v16bf a = load_a_g(arow + k0, K, lane);
#pragma unroll
        for (int c = 0; c < 4; ++c) {
            v16bf b = load_b_g(brow + (size_t)(c * 16) * K + k0, K, lane);
            acc[c] = wmma_bf16(a, b, acc[c]);
        }
    }

    const int col = lane & 15;
    const int off = (lane & 16) ? 8 : 0;
    const int region = n0 / Cc;        // 0=q 1=k 2=v
    const int col0 = n0 % Cc;
    const int h = col0 / Dc;

#pragma unroll
    for (int c = 0; c < 4; ++c) {
        float bias = b_attn[n0 + c * 16 + col];
#pragma unroll
        for (int r = 0; r < 8; ++r) acc[c][r] += bias;
    }

    if (region == 2) {  // V: store transposed [B,NH,D,T]
#pragma unroll
        for (int c = 0; c < 4; ++c) {
            const int dd = c * 16 + col;
#pragma unroll
            for (int r = 0; r < 8; ++r) {
                int m = m0 + r + off;
                int b = m >> 11, t = m & (Tc - 1);
                vtb[(((size_t)b * NHc + h) * Dc + dd) * Tc + t] = (bf16_t)acc[c][r];
            }
        }
    } else {            // Q/K: RoPE. partner of tile c is tile c^2, same lane slot.
        bf16_t* outp = (region == 0) ? qb : kbuf;
#pragma unroll
        for (int c = 0; c < 4; ++c) {
            const int dd = c * 16 + col;
            const float sgn = (c < 2) ? -1.0f : 1.0f;
            const int cp = c ^ 2;
#pragma unroll
            for (int r = 0; r < 8; ++r) {
                int m = m0 + r + off;
                int b = m >> 11, t = m & (Tc - 1);
                float cs = cosT[t * Dc + dd];
                float sn = sinT[t * Dc + dd];
                float v = acc[c][r] * cs + sgn * acc[cp][r] * sn;
                outp[(((size_t)b * NHc + h) * Tc + t) * Dc + dd] = (bf16_t)v;
            }
        }
    }
}

// ---------------- causal flash attention ----------------
// one wave = one 16-row Q tile; K/V streamed in 32-wide chunks; bf16 WMMA, f32 online softmax
__global__ __launch_bounds__(128) void attn_kernel(
    const bf16_t* __restrict__ qb, const bf16_t* __restrict__ kbuf,
    const bf16_t* __restrict__ vtb, bf16_t* __restrict__ yb)
{
    __shared__ __align__(16) bf16_t pls[4][16 * 32];   // per-wave P staging
    const int lane = threadIdx.x & 31;
    const int wave = threadIdx.x >> 5;
    const int q0 = blockIdx.x * 64 + wave * 16;
    const int bh = blockIdx.y;
    const int b = bh / NHc, h = bh % NHc;
    const bf16_t* Qh = qb + (size_t)bh * Tc * Dc;
    const bf16_t* Kh = kbuf + (size_t)bh * Tc * Dc;
    const bf16_t* Vh = vtb + (size_t)bh * Dc * Tc;   // [D, T]
    bf16_t* pw = pls[wave];

    const v16bf qf0 = load_a_g(Qh + (size_t)q0 * Dc + 0, Dc, lane);
    const v16bf qf1 = load_a_g(Qh + (size_t)q0 * Dc + 32, Dc, lane);

    v8f o[4];
#pragma unroll
    for (int dt = 0; dt < 4; ++dt)
#pragma unroll
        for (int i = 0; i < 8; ++i) o[dt][i] = 0.0f;

    float rm[8], rs[8];
#pragma unroll
    for (int r = 0; r < 8; ++r) { rm[r] = -__builtin_inff(); rs[r] = 0.0f; }

    const int col = lane & 15;
    const int off = (lane & 16) ? 8 : 0;
    const float scale = 0.125f;   // 1/sqrt(64)

    for (int n0 = 0; n0 < q0 + 16; n0 += 32) {
        // S = Q @ K^T  (two 16x16 tiles)
        v8f s0, s1;
#pragma unroll
        for (int i = 0; i < 8; ++i) { s0[i] = 0.0f; s1[i] = 0.0f; }
        {
            v16bf kf0a = load_b_g(Kh + (size_t)n0 * Dc + 0, Dc, lane);
            v16bf kf0b = load_b_g(Kh + (size_t)n0 * Dc + 32, Dc, lane);
            s0 = wmma_bf16(qf0, kf0a, s0);
            s0 = wmma_bf16(qf1, kf0b, s0);
            v16bf kf1a = load_b_g(Kh + (size_t)(n0 + 16) * Dc + 0, Dc, lane);
            v16bf kf1b = load_b_g(Kh + (size_t)(n0 + 16) * Dc + 32, Dc, lane);
            s1 = wmma_bf16(qf0, kf1a, s1);
            s1 = wmma_bf16(qf1, kf1b, s1);
        }
        // scale + causal mask
#pragma unroll
        for (int r = 0; r < 8; ++r) {
            const int qi = q0 + r + off;
            float v0 = s0[r] * scale, v1 = s1[r] * scale;
            if (n0 + col > qi)       v0 = -__builtin_inff();
            if (n0 + 16 + col > qi)  v1 = -__builtin_inff();
            s0[r] = v0; s1[r] = v1;
        }
        // online softmax: row = (r, half) — rows live within one 16-lane half
        float scrow[8];
#pragma unroll
        for (int r = 0; r < 8; ++r) {
            float cm = fmaxf(s0[r], s1[r]);
#pragma unroll
            for (int msk = 1; msk < 16; msk <<= 1) cm = fmaxf(cm, __shfl_xor(cm, msk, 32));
            float mn = fmaxf(rm[r], cm);
            float sc = __expf(rm[r] - mn);
            float p0 = __expf(s0[r] - mn);
            float p1 = __expf(s1[r] - mn);
            float ps = p0 + p1;
#pragma unroll
            for (int msk = 1; msk < 16; msk <<= 1) ps += __shfl_xor(ps, msk, 32);
            rs[r] = rs[r] * sc + ps;
            rm[r] = mn;
            s0[r] = p0; s1[r] = p1;
            scrow[r] = sc;
        }
#pragma unroll
        for (int dt = 0; dt < 4; ++dt)
#pragma unroll
            for (int r = 0; r < 8; ++r) o[dt][r] *= scrow[r];

        // re-lay P (C-layout) into A-layout via per-wave LDS
#pragma unroll
        for (int r = 0; r < 8; ++r) {
            pw[(r + off) * 32 + col]      = (bf16_t)s0[r];
            pw[(r + off) * 32 + col + 16] = (bf16_t)s1[r];
        }
        asm volatile("s_wait_dscnt 0" ::: "memory");   // wave-local LDS RAW (cross-lane)
        v16bf pf;
        {
            const int koff = (lane & 16) ? 8 : 0;
            const bf16_t* p = pw + (size_t)(lane & 15) * 32 + koff;
            v8bf lo = *(const v8bf*)(p);
            v8bf hi = *(const v8bf*)(p + 16);
#pragma unroll
            for (int i = 0; i < 8; ++i) { pf[i] = lo[i]; pf[i + 8] = hi[i]; }
        }
        // O += P @ V   (V^T rows are contiguous in t)
#pragma unroll
        for (int dt = 0; dt < 4; ++dt) {
            v16bf vf = load_b_g(Vh + (size_t)(dt * 16) * Tc + n0, Tc, lane);
            o[dt] = wmma_bf16(pf, vf, o[dt]);
        }
    }

    // normalize and write y in [B,T,C] (bf16 for the projection GEMM)
    float inv[8];
#pragma unroll
    for (int r = 0; r < 8; ++r) inv[r] = 1.0f / rs[r];
#pragma unroll
    for (int dt = 0; dt < 4; ++dt) {
#pragma unroll
        for (int r = 0; r < 8; ++r) {
            const int qi = q0 + r + off;
            const int cc = h * Dc + dt * 16 + col;
            yb[((size_t)b * Tc + qi) * Cc + cc] = (bf16_t)(o[dt][r] * inv[r]);
        }
    }
}

// ---------------- output projection: out = y @ w_proj^T + b_proj (fp32 out) ----------------
__global__ __launch_bounds__(128) void proj_kernel(
    const bf16_t* __restrict__ yb, const bf16_t* __restrict__ wp,
    const float* __restrict__ b_proj, float* __restrict__ out)
{
    const int lane = threadIdx.x & 31;
    const int wave = threadIdx.x >> 5;
    const int m0 = blockIdx.y * 64 + wave * 16;
    const int n0 = blockIdx.x * 64;
    const int K = Cc;

    v8f acc[4];
#pragma unroll
    for (int c = 0; c < 4; ++c)
#pragma unroll
        for (int i = 0; i < 8; ++i) acc[c][i] = 0.0f;

    const bf16_t* arow = yb + (size_t)m0 * K;
    const bf16_t* brow = wp + (size_t)n0 * K;
    for (int k0 = 0; k0 < K; k0 += 32) {
        v16bf a = load_a_g(arow + k0, K, lane);
#pragma unroll
        for (int c = 0; c < 4; ++c) {
            v16bf b = load_b_g(brow + (size_t)(c * 16) * K + k0, K, lane);
            acc[c] = wmma_bf16(a, b, acc[c]);
        }
    }

    const int col = lane & 15;
    const int off = (lane & 16) ? 8 : 0;
#pragma unroll
    for (int c = 0; c < 4; ++c) {
        const int n = n0 + c * 16 + col;
        const float bias = b_proj[n];
#pragma unroll
        for (int r = 0; r < 8; ++r) {
            const int m = m0 + r + off;
            out[(size_t)m * Cc + n] = acc[c][r] + bias;
        }
    }
}

// ---------------- launcher ----------------
extern "C" void kernel_launch(void* const* d_in, const int* in_sizes, int n_in,
                              void* d_out, int out_size, void* d_ws, size_t ws_size,
                              hipStream_t stream) {
    const float* x      = (const float*)d_in[0];
    const float* w_attn = (const float*)d_in[1];
    const float* b_attn = (const float*)d_in[2];
    const float* w_proj = (const float*)d_in[3];
    const float* b_proj = (const float*)d_in[4];
    const float* cosT   = (const float*)d_in[5];
    const float* sinT   = (const float*)d_in[6];
    float* out = (float*)d_out;

    char* ws = (char*)d_ws;
    size_t off = 0;
    auto alloc = [&](size_t bytes) -> void* {
        void* p = ws + off;
        off += (bytes + 255) & ~(size_t)255;
        return p;
    };
    const size_t NE = (size_t)Bc * Tc * Cc;          // 8,388,608
    bf16_t* xb  = (bf16_t*)alloc(NE * 2);
    bf16_t* wab = (bf16_t*)alloc((size_t)3 * Cc * Cc * 2);
    bf16_t* wpb = (bf16_t*)alloc((size_t)Cc * Cc * 2);
    bf16_t* qb  = (bf16_t*)alloc(NE * 2);
    bf16_t* kb  = (bf16_t*)alloc(NE * 2);
    bf16_t* vtb = (bf16_t*)alloc(NE * 2);
    bf16_t* yb  = (bf16_t*)alloc(NE * 2);

    {
        int n = (int)NE;
        cvt_bf16_kernel<<<(n + 255) / 256, 256, 0, stream>>>(x, xb, n);
    }
    {
        int n = 3 * Cc * Cc;
        cvt_bf16_kernel<<<(n + 255) / 256, 256, 0, stream>>>(w_attn, wab, n);
    }
    {
        int n = Cc * Cc;
        cvt_bf16_kernel<<<(n + 255) / 256, 256, 0, stream>>>(w_proj, wpb, n);
    }

    // QKV + RoPE: M=B*T=8192 (128 tiles of 64), N=3C=3072 (48 tiles of 64)
    qkv_rope_kernel<<<dim3(48, 128), 128, 0, stream>>>(xb, wab, b_attn, cosT, sinT, qb, kb, vtb);

    // flash attention: (T/64=32 q-tiles, B*NH=64 heads)
    attn_kernel<<<dim3(32, 64), 128, 0, stream>>>(qb, kb, vtb, yb);

    // projection: N=C=1024 (16 tiles), M=8192 (128 tiles)
    proj_kernel<<<dim3(16, 128), 128, 0, stream>>>(yb, wpb, b_proj, out);
}
